// StructuredTransform_36524401885691
// MI455X (gfx1250) — matrix operand order repaired and checked by reading, not compiled, not run on hardware
//
#include <hip/hip_runtime.h>

// ---- CDNA5 wave32 WMMA types ----
typedef __attribute__((ext_vector_type(16))) _Float16 v16h;
typedef __attribute__((ext_vector_type(8)))  _Float16 v8h;
typedef __attribute__((ext_vector_type(8)))  float    v8f;

#define N_RES 4096
#define N_IN  4096
#define NPAD  8192   // 64 * 128
#define BATCH 4096
#define ROWS_PER_BLOCK 2
#define XT_P  72     // pitch (halves) for XT[128][64]: 144B rows, 16B aligned
#define TH_P  136    // pitch (halves) for TH/TL[64][128]: 272B rows, 16B aligned

__device__ __forceinline__ v8f wmma_f16(v16h a, v16h b, v8f c) {
  // D = A(16x32 f16) * B(32x16 f16) + C(16x16 f32)
  return __builtin_amdgcn_wmma_f32_16x16x32_f16(
      /*neg_a=*/false, a, /*neg_b=*/false, b,
      /*c_mod=*/(short)0, c, /*reuse_a=*/false, /*reuse_b=*/false);
}

__device__ __forceinline__ v16h join16(v8h lo, v8h hi) {
  return __builtin_shufflevector(lo, hi, 0,1,2,3,4,5,6,7,8,9,10,11,12,13,14,15);
}
// 16 contiguous halves (B-operand run): p[0..15]
__device__ __forceinline__ v16h ld16_contig(const _Float16* p) {
  return join16(*(const v8h*)p, *(const v8h*)(p + 8));
}
// A-operand runs: p[0..7] and p[16..23]
__device__ __forceinline__ v16h ld16_gap(const _Float16* p) {
  return join16(*(const v8h*)p, *(const v8h*)(p + 16));
}

// A-operand fragment of a Hadamard matrix: A[m][k] = (-1)^popc(m&k).
// 16-bit A 16x32 fragment in contiguous k order (as pass-A B uses): lane L -> M = mBase + (L&15);
// element e -> K = kBase + 16*(L>=16) + e.
__device__ __forceinline__ v16h had_A_frag(int mBase, int kBase, int lane) {
  const int m = mBase + (lane & 15);
  const int khalf = (lane >> 4) << 4;
  v16h a;
#pragma unroll
  for (int e = 0; e < 16; ++e) {
    const int k = kBase + khalf + e;
    a[e] = (__popc((unsigned)(m & k)) & 1) ? (_Float16)(-1.0f) : (_Float16)(1.0f);
  }
  return a;
}

// B-operand fragment of a Hadamard matrix: B[k][n] = (-1)^popc(k&n).
// CDNA5 16-bit B 32x16 layout: lane L -> N = nBase + (L&15);
// element e -> K = kBase + 8*(L>=16) + (e&7) + 16*(e>>3).
__device__ __forceinline__ v16h had_B_frag(int kBase, int nBase, int lane) {
  const int n = nBase + (lane & 15);
  const int kh = (lane >> 4) << 3;
  v16h b;
#pragma unroll
  for (int e = 0; e < 16; ++e) {
    const int k = kBase + kh + (e & 7) + ((e >> 3) << 4);
    b[e] = (__popc((unsigned)(k & n)) & 1) ? (_Float16)(-1.0f) : (_Float16)(1.0f);
  }
  return b;
}

__global__ __launch_bounds__(256)
void reservoir_fwht_wmma(const float* __restrict__ state,
                         const float* __restrict__ inputs,
                         const float* __restrict__ d1,
                         const float* __restrict__ d2,
                         const float* __restrict__ d3,
                         const float* __restrict__ bias,
                         float* __restrict__ out) {
  __shared__ _Float16 XTh[128 * XT_P];  // X^T hi (f16)  18.0 KB
  __shared__ _Float16 XTl[128 * XT_P];  // X^T lo (f16)  18.0 KB
  __shared__ _Float16 TH [64  * TH_P];  // (H64*X) hi    17.0 KB
  __shared__ _Float16 TL [64  * TH_P];  // (H64*X) lo    17.0 KB

  const int tid  = threadIdx.x;
  const int wave = tid >> 5;        // 8 waves (wave32)
  const int lane = tid & 31;

  // ---- persistent Hadamard operand fragments (constant across rows/rounds) ----
  // Pass A (T = H64 * X): wave owns row-tile itA, column tiles jtA0..jtA0+3.
  const int itA  = wave & 3;
  const int jtA0 = (wave >> 2) * 4;
  const v16h A64_0 = had_A_frag(itA * 16, 0,  lane);
  const v16h A64_1 = had_A_frag(itA * 16, 32, lane);
  // Pass B (Y = T * H128): wave owns column tile ct, loops row tiles.
  const int ct = wave;
  v16h B128[4];
#pragma unroll
  for (int ks = 0; ks < 4; ++ks) B128[ks] = had_B_frag(ks * 32, ct * 16, lane);

  for (int rr = 0; rr < ROWS_PER_BLOCK; ++rr) {
    const int row = blockIdx.x * ROWS_PER_BLOCK + rr;
    const float* srow = state  + (size_t)row * N_RES;
    const float* irow = inputs + (size_t)row * N_IN;

    // Prefetch next row's global data while this row computes (global_prefetch_b8).
    if (rr + 1 < ROWS_PER_BLOCK) {
      const float* ns = srow + N_RES;
      const float* ni = irow + N_IN;
      __builtin_prefetch(ns + tid * 16, 0, 3);   // 256 thr * 64B = 16KB
      __builtin_prefetch(ni + tid * 16, 0, 3);
    }

    // ---- stage 0: x = concat(0.9*state, 0.4*inputs) * d1, f32 -> f16 hi/lo,
    //      stored transposed: XT[jl][jh] = x[jh*128 + jl].
    //      Two uniform loops (no per-lane select between state/inputs). ----
#pragma unroll 4
    for (int j = tid; j < N_RES; j += 256) {
      const float v = 0.9f * srow[j] * d1[j];
      const _Float16 h = (_Float16)v;
      const int jl = j & 127, jh = j >> 7;
      XTh[jl * XT_P + jh] = h;
      XTl[jl * XT_P + jh] = (_Float16)(v - (float)h);
    }
#pragma unroll 4
    for (int j = tid; j < N_IN; j += 256) {
      const int jj = j + N_RES;
      const float v = 0.4f * irow[j] * d1[jj];
      const _Float16 h = (_Float16)v;
      const int jl = jj & 127, jh = jj >> 7;
      XTh[jl * XT_P + jh] = h;
      XTl[jl * XT_P + jh] = (_Float16)(v - (float)h);
    }
    __syncthreads();

    // ---- three FWHT rounds: Y = H64 * X * H128 ----
#pragma unroll
    for (int r = 0; r < 3; ++r) {
      const bool last = (r == 2);

      // ============ Pass A: T = H64 * X (split hi/lo at write) ============
      // Last round only consumes T rows 0..31 -> waves owning row-tiles 2,3 skip.
      if (!(last && itA >= 2)) {
#pragma unroll
        for (int jj = 0; jj < 4; ++jj) {
          const int jt = jtA0 + jj;
          const int n  = jt * 16 + (lane & 15);          // output column (jl)
          v8f c = {};
#pragma unroll
          for (int ks = 0; ks < 2; ++ks) {
            const int kb = ks * 32 + ((lane >> 4) << 4); // K = jh block
            const v16h bh = ld16_contig(&XTh[n * XT_P + kb]);
            const v16h bl = ld16_contig(&XTl[n * XT_P + kb]);
            const v16h A  = ks ? A64_1 : A64_0;
            c = wmma_f16(A, bh, c);
            c = wmma_f16(A, bl, c);
          }
          // D layout: VGPR e -> M = e + 8*(lane>=16), N = lane&15
          const int mrow = itA * 16 + ((lane >> 4) << 3);
#pragma unroll
          for (int e = 0; e < 8; ++e) {
            const float f = c[e];
            const _Float16 h = (_Float16)f;
            TH[(mrow + e) * TH_P + n] = h;
            TL[(mrow + e) * TH_P + n] = (_Float16)(f - (float)h);
          }
        }
      }
      __syncthreads();

      // ============ Pass B: Y = T * H128 ============
      const int itMax = last ? 2 : 4;                 // last round: only k < 4096
      const float* dnx = (r == 0) ? d2 : d3;

#pragma unroll
      for (int it = 0; it < 4; ++it) {
        if (it >= itMax) break;
        const int m = it * 16 + (lane & 15);          // A row (kh)
        v8f c = {};
#pragma unroll
        for (int ks = 0; ks < 4; ++ks) {
          const int kb = ks * 32 + ((lane >> 4) << 3); // A K-pattern base
          const v16h ah = ld16_gap(&TH[m * TH_P + kb]);
          const v16h al = ld16_gap(&TL[m * TH_P + kb]);
          c = wmma_f16(ah, B128[ks], c);
          c = wmma_f16(al, B128[ks], c);
        }

        const int mrow = it * 16 + ((lane >> 4) << 3);
        const int ncol = ct * 16 + (lane & 15);
        if (!last) {
          // X_next^T[jl][jh] = (Y/64) * d_next, split hi/lo
#pragma unroll
          for (int e = 0; e < 8; ++e) {
            const int kh = mrow + e;
            const int j  = kh * 128 + ncol;
            const float f = c[e] * (1.0f / 64.0f) * dnx[j];
            const _Float16 h = (_Float16)f;
            XTh[ncol * XT_P + kh] = h;
            XTl[ncol * XT_P + kh] = (_Float16)(f - (float)h);
          }
        } else {
          // total scale so far 1/4096; remaining 1/2 => /8192, + bias, erf
#pragma unroll
          for (int e = 0; e < 8; ++e) {
            const int k = (mrow + e) * 128 + ncol;    // < 4096 (it < 2)
            const float f = c[e] * 0.5f + bias[k];
            out[(size_t)row * N_RES + k] = erff(f);
          }
        }
      }
      __syncthreads();
    }
  }
}

extern "C" void kernel_launch(void* const* d_in, const int* in_sizes, int n_in,
                              void* d_out, int out_size, void* d_ws, size_t ws_size,
                              hipStream_t stream) {
  (void)in_sizes; (void)n_in; (void)out_size; (void)d_ws; (void)ws_size;
  const float* state  = (const float*)d_in[0];
  const float* inputs = (const float*)d_in[1];
  const float* d1     = (const float*)d_in[2];
  const float* d2     = (const float*)d_in[3];
  const float* d3     = (const float*)d_in[4];
  const float* bias   = (const float*)d_in[5];
  float* out = (float*)d_out;

  dim3 grid(BATCH / ROWS_PER_BLOCK), block(256);
  hipLaunchKernelGGL(reservoir_fwht_wmma, grid, block, 0, stream,
                     state, inputs, d1, d2, d3, bias, out);
}
